// MultiHeadAttention_62912680951955
// MI455X (gfx1250) — matrix operand order repaired and checked by reading, not compiled, not run on hardware
//
#include <hip/hip_runtime.h>
#include <hip/hip_bf16.h>

typedef __attribute__((ext_vector_type(16))) _Float16 v16h;
typedef __attribute__((ext_vector_type(8)))  _Float16 v8h;
typedef __attribute__((ext_vector_type(8)))  float    v8f;
typedef __attribute__((ext_vector_type(4)))  int      i32x4;

#define DMODEL 1024
#define SEQ    2048
#define NH     16
#define DH     64
#define MROWS  4096   // B*S = 2*2048

#define GLOBAL_AS __attribute__((address_space(1)))
#define LDS_AS    __attribute__((address_space(3)))

// ---------------------------------------------------------------------------
// CDNA5 helpers: async global->LDS copy (ASYNCcnt) and DPP 16-lane butterflies
// ---------------------------------------------------------------------------
__device__ inline void async_b128(const _Float16* g, const _Float16* l) {
#if __has_builtin(__builtin_amdgcn_global_load_async_to_lds_b128)
    __builtin_amdgcn_global_load_async_to_lds_b128(
        (GLOBAL_AS i32x4*)(uintptr_t)g,
        (LDS_AS i32x4*)(unsigned)(uintptr_t)l, 0, 0);
#else
    unsigned lo = (unsigned)(uintptr_t)l;
    asm volatile("global_load_async_to_lds_b128 %0, %1, off"
                 :: "v"(lo), "v"((uintptr_t)g) : "memory");
#endif
}

__device__ inline void wait_async0() {
#if __has_builtin(__builtin_amdgcn_s_wait_asynccnt)
    __builtin_amdgcn_s_wait_asynccnt(0);
#else
    asm volatile("s_wait_asynccnt 0x0" ::: "memory");
#endif
}

// max/sum all-reduce across each 16-lane half (wave32) — DPP ROW_XMASK, no LDS.
// EXEC is all-ones in these kernels, so old=0 + bound_ctrl=1 is safe and avoids
// the pre-copy mov that the old-preserving form requires.
__device__ inline float red_max16(float v) {
#if __has_builtin(__builtin_amdgcn_update_dpp)
    int t;
    t = __builtin_amdgcn_update_dpp(0, __float_as_int(v), 0x161, 0xf, 0xf, true);
    v = fmaxf(v, __int_as_float(t));
    t = __builtin_amdgcn_update_dpp(0, __float_as_int(v), 0x162, 0xf, 0xf, true);
    v = fmaxf(v, __int_as_float(t));
    t = __builtin_amdgcn_update_dpp(0, __float_as_int(v), 0x164, 0xf, 0xf, true);
    v = fmaxf(v, __int_as_float(t));
    t = __builtin_amdgcn_update_dpp(0, __float_as_int(v), 0x168, 0xf, 0xf, true);
    v = fmaxf(v, __int_as_float(t));
    return v;
#else
    v = fmaxf(v, __shfl_xor(v, 1, 16));
    v = fmaxf(v, __shfl_xor(v, 2, 16));
    v = fmaxf(v, __shfl_xor(v, 4, 16));
    v = fmaxf(v, __shfl_xor(v, 8, 16));
    return v;
#endif
}

__device__ inline float red_sum16(float v) {
#if __has_builtin(__builtin_amdgcn_update_dpp)
    int t;
    t = __builtin_amdgcn_update_dpp(0, __float_as_int(v), 0x161, 0xf, 0xf, true);
    v += __int_as_float(t);
    t = __builtin_amdgcn_update_dpp(0, __float_as_int(v), 0x162, 0xf, 0xf, true);
    v += __int_as_float(t);
    t = __builtin_amdgcn_update_dpp(0, __float_as_int(v), 0x164, 0xf, 0xf, true);
    v += __int_as_float(t);
    t = __builtin_amdgcn_update_dpp(0, __float_as_int(v), 0x168, 0xf, 0xf, true);
    v += __int_as_float(t);
    return v;
#else
    v += __shfl_xor(v, 1, 16);
    v += __shfl_xor(v, 2, 16);
    v += __shfl_xor(v, 4, 16);
    v += __shfl_xor(v, 8, 16);
    return v;
#endif
}

// ---------------------------------------------------------------------------
// Weight pre-pack: f32 row-major [D][D] -> f16 WMMA B-fragment tile order.
// ---------------------------------------------------------------------------
__global__ void mha_pack_w(const float* __restrict__ W, _Float16* __restrict__ Wp) {
    int idx = blockIdx.x * blockDim.x + threadIdx.x;
    if (idx >= DMODEL * DMODEL) return;
    int k = idx >> 10;
    int n = idx & 1023;
    int kt = k >> 5, kk = k & 31;
    int nt = n >> 4, nn = n & 15;
    int half = (kk >> 3) & 1, elem = (kk & 7) | ((kk >> 4) << 3);
    int lane = nn + (half << 4);
    size_t dst = ((size_t)(kt * (DMODEL >> 4) + nt) * 32 + lane) * 16 + elem;
    Wp[dst] = (_Float16)W[idx];
}

__global__ void mha_cvt_f16(const float* __restrict__ x, _Float16* __restrict__ y, int n) {
    int i = blockIdx.x * blockDim.x + threadIdx.x;
    if (i < n) y[i] = (_Float16)x[i];
}

// V row-major [4096][1024] -> Vt[b][h][dh][SEQ]
__global__ void mha_vtrans(const _Float16* __restrict__ Vr, _Float16* __restrict__ Vt) {
    int idx = blockIdx.x * blockDim.x + threadIdx.x;
    if (idx >= MROWS * DMODEL) return;
    int row = idx >> 10, col = idx & 1023;
    int b = row >> 11, pos = row & 2047;
    int h = col >> 6,  d   = col & 63;
    Vt[(((size_t)((b * NH + h) * DH + d)) << 11) + pos] = Vr[idx];
}

// ---------------------------------------------------------------------------
// NN GEMM: C[M,N] = A[M,K] (row-major f16) x Wp (packed f16) + bias.
// One wave -> 16x64 output tile.
// ---------------------------------------------------------------------------
template <typename OUT>
__global__ void mha_gemm16(const _Float16* __restrict__ A, const _Float16* __restrict__ Bp,
                           const float* __restrict__ bias, OUT* __restrict__ C,
                           int M, int N, int K) {
    int lane = threadIdx.x & 31;
    int wid  = (blockIdx.x * blockDim.x + threadIdx.x) >> 5;
    int numNT = N >> 6;
    int mt = wid / numNT, nt = wid % numNT;
    if (mt >= (M >> 4)) return;
    int m0  = mt << 4;
    int row = m0 + (lane & 15);
    int ka  = (lane >> 4) << 3;
    int ntiles16 = N >> 4;

    v8f acc[4] = {};
    for (int kt = 0; kt < (K >> 5); ++kt) {
        const _Float16* pa = A + (size_t)row * K + kt * 32 + ka;
        v8h lo = *(const v8h*)pa;
        v8h hi = *(const v8h*)(pa + 16);
        v16h a;
#pragma unroll
        for (int i = 0; i < 8; ++i) { a[i] = lo[i]; a[8 + i] = hi[i]; }
        v16h bf[4];
#pragma unroll
        for (int j = 0; j < 4; ++j) {
            size_t bb = ((size_t)(kt * ntiles16 + (nt * 4 + j)) * 32 + lane) * 16;
            bf[j] = *(const v16h*)(Bp + bb);
        }
#pragma unroll
        for (int j = 0; j < 4; ++j)
            acc[j] = __builtin_amdgcn_wmma_f32_16x16x32_f16(
                false, a, false, bf[j], (short)0, acc[j], false, false);
    }
    int hh = lane >> 4;
#pragma unroll
    for (int j = 0; j < 4; ++j) {
        int n = (nt * 4 + j) * 16 + (lane & 15);
        float bv = bias[n];
#pragma unroll
        for (int r = 0; r < 8; ++r)
            C[(size_t)(m0 + r + 8 * hh) * N + n] = (OUT)(acc[j][r] + bv);
    }
}

// ---------------------------------------------------------------------------
// Flash attention. Block = 4 waves, all on the same (b,h) with 4 consecutive
// 16-query tiles. K/V 32-key tiles are double-buffered in LDS via
// GLOBAL_LOAD_ASYNC_TO_LDS_B128 (ASYNCcnt), shared by the 4 waves; the next
// tile's copy overlaps the current tile's wmma+softmax.
// ---------------------------------------------------------------------------
__global__ void __launch_bounds__(128)
mha_attn(const _Float16* __restrict__ Q, const _Float16* __restrict__ Km,
         const _Float16* __restrict__ Vt, _Float16* __restrict__ Ctx) {
    __shared__ _Float16 sK[2][32][64];   // [buf][key][d]     4KB each
    __shared__ _Float16 sV[2][64][32];   // [buf][d][key]     4KB each
    __shared__ _Float16 sP[4][16][32];   // per-wave P staging

    int tid  = threadIdx.x;
    int lane = tid & 31;
    int wv   = tid >> 5;
    int bid  = blockIdx.x;               // 2*16*32 = 1024 blocks
    int qg = bid & 31;
    int h  = (bid >> 5) & 15;
    int b  = bid >> 9;
    int qt = qg * 4 + wv;

    int m0 = b * SEQ + qt * 16;
    int hh = lane >> 4;
    int ln = lane & 15;

    const _Float16* Kg = Km + (size_t)(b * SEQ) * DMODEL + h * DH;     // [pos][dh]
    const _Float16* Vg = Vt + ((size_t)((b * NH + h) * DH) << 11);     // [dh][pos]

    // Q A-fragments (two K=32 steps over dh=64)
    v16h aq[2];
#pragma unroll
    for (int s = 0; s < 2; ++s) {
        const _Float16* p = Q + (size_t)(m0 + ln) * DMODEL + h * DH + s * 32 + (hh << 4);
        v8h lo = *(const v8h*)p;
        v8h hi = *(const v8h*)(p + 8);
#pragma unroll
        for (int i = 0; i < 8; ++i) { aq[s][i] = lo[i]; aq[s][8 + i] = hi[i]; }
    }

    float mrun[8], lrun[8];
#pragma unroll
    for (int r = 0; r < 8; ++r) { mrun[r] = -3.0e38f; lrun[r] = 0.f; }
    v8f acc[4] = {};
    _Float16* sp = &sP[wv][0][0];

    // async stage of one 32-key K/V tile: 128 threads x 2 b128 chunks each
    auto stage = [&](int kb, int buf) {
#pragma unroll
        for (int c = 0; c < 2; ++c) {
            int cid = tid * 2 + c;                       // 0..255
            int key = cid >> 3, ko = (cid & 7) << 3;     // 8 chunks per key row
            async_b128(Kg + (size_t)(kb + key) * DMODEL + ko, &sK[buf][key][ko]);
            int d = cid >> 2, vo = (cid & 3) << 3;       // 4 chunks per d row
            async_b128(Vg + (size_t)d * SEQ + kb + vo, &sV[buf][d][vo]);
        }
    };

    stage(0, 0);
    wait_async0();
    __syncthreads();
    int cur = 0;

    for (int kb = 0; kb < SEQ; kb += 32) {
        if (kb + 32 < SEQ) stage(kb + 32, cur ^ 1);

        // logits: load all 4 B-fragments first, then 2 key tiles x 2 K-steps
        v16h bk[2][2];
#pragma unroll
        for (int t = 0; t < 2; ++t)
#pragma unroll
            for (int s = 0; s < 2; ++s)
                bk[t][s] = *(const v16h*)&sK[cur][t * 16 + ln][s * 32 + (hh << 4)];
        v8f sfr[2] = {};
#pragma unroll
        for (int t = 0; t < 2; ++t)
#pragma unroll
            for (int s = 0; s < 2; ++s)
                sfr[t] = __builtin_amdgcn_wmma_f32_16x16x32_f16(
                    false, aq[s], false, bk[t][s], (short)0, sfr[t], false, false);

        // online softmax; cross-lane reductions via DPP ROW_XMASK
#pragma unroll
        for (int r = 0; r < 8; ++r) {
            float s0 = sfr[0][r] * 0.125f;   // 1/sqrt(64)
            float s1 = sfr[1][r] * 0.125f;
            float v = red_max16(fmaxf(s0, s1));
            float mnew = fmaxf(mrun[r], v);
            float corr = __expf(mrun[r] - mnew);
            mrun[r] = mnew;
            float p0 = __expf(s0 - mnew);
            float p1 = __expf(s1 - mnew);
            lrun[r] = lrun[r] * corr + red_sum16(p0 + p1);
#pragma unroll
            for (int j = 0; j < 4; ++j) acc[j][r] *= corr;
            int mrow = r + 8 * hh;
            sp[mrow * 32 + ln]      = (_Float16)p0;
            sp[mrow * 32 + 16 + ln] = (_Float16)p1;
        }
        asm volatile("s_wait_dscnt 0" ::: "memory");

        // P as A-fragment (16x32) + all 4 V B-fragments, then the wmma chain
        v16h ap;
        {
            const _Float16* p = sp + ln * 32 + (hh << 4);
            v8h lo = *(const v8h*)p;
            v8h hi = *(const v8h*)(p + 8);
#pragma unroll
            for (int i = 0; i < 8; ++i) { ap[i] = lo[i]; ap[8 + i] = hi[i]; }
        }
        v16h bv[4];
#pragma unroll
        for (int j = 0; j < 4; ++j)
            bv[j] = *(const v16h*)&sV[cur][j * 16 + ln][hh << 4];
#pragma unroll
        for (int j = 0; j < 4; ++j)
            acc[j] = __builtin_amdgcn_wmma_f32_16x16x32_f16(
                false, ap, false, bv[j], (short)0, acc[j], false, false);

        wait_async0();      // prefetched tile complete (per-wave ASYNCcnt)
        __syncthreads();    // all waves done reading cur, next buffer visible
        cur ^= 1;
    }

#pragma unroll
    for (int r = 0; r < 8; ++r) {
        float inv = 1.0f / lrun[r];
        int grow = m0 + r + 8 * hh;
#pragma unroll
        for (int j = 0; j < 4; ++j)
            Ctx[(size_t)grow * DMODEL + h * DH + j * 16 + ln] =
                (_Float16)(acc[j][r] * inv);
    }
}

// ---------------------------------------------------------------------------
extern "C" void kernel_launch(void* const* d_in, const int* in_sizes, int n_in,
                              void* d_out, int out_size, void* d_ws, size_t ws_size,
                              hipStream_t stream) {
    const float* q  = (const float*)d_in[0];
    const float* k  = (const float*)d_in[1];
    const float* v  = (const float*)d_in[2];
    const float* Wq = (const float*)d_in[3];
    const float* bq = (const float*)d_in[4];
    const float* Wk = (const float*)d_in[5];
    const float* bk = (const float*)d_in[6];
    const float* Wv = (const float*)d_in[7];
    const float* bv = (const float*)d_in[8];
    const float* Wo = (const float*)d_in[9];
    const float* bo = (const float*)d_in[10];
    float* out = (float*)d_out;

    char* ws = (char*)d_ws;
    const size_t SZ_X = (size_t)MROWS * DMODEL * sizeof(_Float16);   // 8 MB
    const size_t SZ_W = (size_t)DMODEL * DMODEL * sizeof(_Float16);  // 2 MB
    _Float16* Xq16 = (_Float16*)(ws);
    _Float16* Xk16 = (_Float16*)(ws + SZ_X);
    _Float16* Xv16 = (_Float16*)(ws + 2 * SZ_X);
    _Float16* Wqp  = (_Float16*)(ws + 3 * SZ_X);
    _Float16* Wkp  = (_Float16*)(ws + 3 * SZ_X + SZ_W);
    _Float16* Wvp  = (_Float16*)(ws + 3 * SZ_X + 2 * SZ_W);
    _Float16* Wop  = (_Float16*)(ws + 3 * SZ_X + 3 * SZ_W);
    _Float16* Q16  = (_Float16*)(ws + 3 * SZ_X + 4 * SZ_W);
    _Float16* K16  = (_Float16*)(ws + 4 * SZ_X + 4 * SZ_W);
    _Float16* V16  = (_Float16*)(ws + 5 * SZ_X + 4 * SZ_W);
    _Float16* Vt   = Xk16;   // reuse: Xk consumed before Vt written
    _Float16* Ctx  = Xq16;   // reuse: Xq consumed before Ctx written

    const int NE = MROWS * DMODEL;
    const int TPB = 256;

    mha_pack_w<<<(DMODEL * DMODEL + TPB - 1) / TPB, TPB, 0, stream>>>(Wq, Wqp);
    mha_pack_w<<<(DMODEL * DMODEL + TPB - 1) / TPB, TPB, 0, stream>>>(Wk, Wkp);
    mha_pack_w<<<(DMODEL * DMODEL + TPB - 1) / TPB, TPB, 0, stream>>>(Wv, Wvp);
    mha_pack_w<<<(DMODEL * DMODEL + TPB - 1) / TPB, TPB, 0, stream>>>(Wo, Wop);
    mha_cvt_f16<<<(NE + TPB - 1) / TPB, TPB, 0, stream>>>(q, Xq16, NE);
    mha_cvt_f16<<<(NE + TPB - 1) / TPB, TPB, 0, stream>>>(k, Xk16, NE);
    mha_cvt_f16<<<(NE + TPB - 1) / TPB, TPB, 0, stream>>>(v, Xv16, NE);

    int gemm_blocks = ((MROWS >> 4) * (DMODEL >> 6) * 32) / TPB;
    mha_gemm16<_Float16><<<gemm_blocks, TPB, 0, stream>>>(Xq16, Wqp, bq, Q16,
                                                          MROWS, DMODEL, DMODEL);
    mha_gemm16<_Float16><<<gemm_blocks, TPB, 0, stream>>>(Xk16, Wkp, bk, K16,
                                                          MROWS, DMODEL, DMODEL);
    mha_gemm16<_Float16><<<gemm_blocks, TPB, 0, stream>>>(Xv16, Wvp, bv, V16,
                                                          MROWS, DMODEL, DMODEL);

    mha_vtrans<<<(NE + TPB - 1) / TPB, TPB, 0, stream>>>(V16, Vt);

    mha_attn<<<1024, 128, 0, stream>>>(Q16, K16, Vt, Ctx);

    mha_gemm16<float><<<gemm_blocks, TPB, 0, stream>>>(Ctx, Wop, bo, out,
                                                       MROWS, DMODEL, DMODEL);
}
